// VSSBlock_82789789598503
// MI455X (gfx1250) — matrix operand order repaired and checked by reading, not compiled, not run on hardware
//
#include <hip/hip_runtime.h>
#include <hip/hip_bf16.h>
#include <math.h>

// ---------------- dimensions ----------------
#define BATCH   2
#define CMODEL  96
#define HW      4096          // 64*64
#define NTOK    (BATCH*HW)    // 8192
#define DINNER  192
#define DSTATE  16
#define NHEADS  16
#define HEADDIM 12
#define DCONV   4
#define CONVDIM 224           // DINNER + 2*DSTATE
#define DPROJ   432           // 2*DINNER + 2*DSTATE + NHEADS
#define LC      64            // scan chunk length (4096 % 64 == 0, batch aligned)
#define NCHUNK  (NTOK/LC)     // 128

typedef __attribute__((ext_vector_type(16))) _Float16 v16h;
typedef __attribute__((ext_vector_type(8)))  float    v8f;

// ---------------- WMMA fragment loaders (ISA 7.12.2, 16-bit layouts) ----------------
// A (16x32 f16): lanes 0-15 rows 0-15 K=[0..7,16..23]; lanes 16-31 same rows K=[8..15,24..31]
__device__ __forceinline__ v16h load_a_frag(const float* __restrict__ A, int lda,
                                            int row, int k0, int lane) {
  const float* p = A + row * lda;
  int base = k0 + ((lane & 16) ? 8 : 0);
  v16h a;
#pragma unroll
  for (int e = 0; e < 8; ++e)  a[e] = (_Float16)p[base + e];
#pragma unroll
  for (int e = 8; e < 16; ++e) a[e] = (_Float16)p[base + 8 + e];
  return a;
}
// B (32x16 f16), W row-major (N x K) so B[k][n] = W[n][k]:
// lanes 0-15 col n, K=[0..7,16..23] ; lanes 16-31 col n, K=[8..15,24..31]
__device__ __forceinline__ v16h load_b_frag(const float* __restrict__ W, int ldw,
                                            int col, int k0, int lane) {
  const float* p = W + col * ldw + k0 + ((lane & 16) ? 8 : 0);
  v16h b;
#pragma unroll
  for (int e = 0; e < 16; ++e) b[e] = (_Float16)p[e + ((e >= 8) ? 8 : 0)];
  return b;
}

#define WMMA_F16(a, b, c) \
  __builtin_amdgcn_wmma_f32_16x16x32_f16(false, (a), false, (b), (short)0, (c), false, false)

// ---------------- generic GEMM: C[m][n] = sum_k A[m][k] * W[n][k] ----------------
// A: M x K row-major f32; W: N x K row-major f32; C: M x N f32.
// One 64x16 output supertile per wave: one B fragment feeds 4 back-to-back WMMAs
// (independent accumulators -> no D->A/B RAW hazard stalls on the matrix pipe).
__global__ void gemm_wmma(const float* __restrict__ A, const float* __restrict__ W,
                          float* __restrict__ C, int M, int N, int K) {
  int lane = threadIdx.x & 31;
  int wave = blockIdx.x * (blockDim.x >> 5) + (threadIdx.x >> 5);
  int ntn = N >> 4;
  int ntiles = (M >> 6) * ntn;          // 64-row supertiles
  if (wave >= ntiles) return;
  int tm = wave / ntn, tn = wave % ntn;
  int m0 = tm << 6, n0 = tn << 4;
  int r0 = m0 + (lane & 15);
  int col = n0 + (lane & 15);
  v8f acc0 = {}, acc1 = {}, acc2 = {}, acc3 = {};
  for (int k0 = 0; k0 < K; k0 += 32) {
    if (k0 + 32 < K)  // hint next K-slab toward the caches
      __builtin_prefetch(A + r0 * K + k0 + 32, 0, 1);
    v16h b  = load_b_frag(W, K, col, k0, lane);
    v16h a0 = load_a_frag(A, K, r0,      k0, lane);
    v16h a1 = load_a_frag(A, K, r0 + 16, k0, lane);
    v16h a2 = load_a_frag(A, K, r0 + 32, k0, lane);
    v16h a3 = load_a_frag(A, K, r0 + 48, k0, lane);
    acc0 = WMMA_F16(a0, b, acc0);
    acc1 = WMMA_F16(a1, b, acc1);
    acc2 = WMMA_F16(a2, b, acc2);
    acc3 = WMMA_F16(a3, b, acc3);
  }
  int roff = (lane & 16) ? 8 : 0;       // D layout: VGPR r -> M = r (+8 upper half-wave)
  float* Cb = C + col;
#pragma unroll
  for (int r = 0; r < 8; ++r) {
    Cb[(m0 + roff + r) * N]      = acc0[r];
    Cb[(m0 + 16 + roff + r) * N] = acc1[r];
    Cb[(m0 + 32 + roff + r) * N] = acc2[r];
    Cb[(m0 + 48 + roff + r) * N] = acc3[r];
  }
}

// ---------------- final conv GEMM with reshape semantics + bias + residual ----------------
// out[b,n,s] = sum_{c<96} Wc[n][c]*lnA[b][c*HW+s] + sum_{c<96} Wc[n][96+c]*lnB[b][c*HW+s]
//              + bias[n] + resid[b,n,s]
__device__ __forceinline__ v16h load_a_strided(const float* __restrict__ A0,
                                               const float* __restrict__ A1,
                                               int k0, int lane) {
  int base = k0 + ((lane & 16) ? 8 : 0);
  v16h a;
#pragma unroll
  for (int e = 0; e < 16; ++e) {
    int k = base + e + ((e >= 8) ? 8 : 0);
    float v = (k < CMODEL) ? A0[k * HW] : A1[(k - CMODEL) * HW];
    a[e] = (_Float16)v;
  }
  return a;
}

__global__ void gemm_final_wmma(const float* __restrict__ lnA, const float* __restrict__ lnB,
                                const float* __restrict__ Wc, const float* __restrict__ bias,
                                const float* __restrict__ resid, float* __restrict__ out) {
  const int M = NTOK, N = CMODEL, K = 2 * CMODEL;
  int lane = threadIdx.x & 31;
  int wave = blockIdx.x * (blockDim.x >> 5) + (threadIdx.x >> 5);
  int ntn = N >> 4;
  int ntiles = (M >> 6) * ntn;
  if (wave >= ntiles) return;
  int tm = wave / ntn, tn = wave % ntn;
  int m0 = tm << 6, n0 = tn << 4;
  int col = n0 + (lane & 15);
  // rows of a 64-row supertile never straddle the batch boundary (4096 % 64 == 0)
  int b = m0 >> 12;
  const float* A0 = lnA + b * (CMODEL * HW);
  const float* A1 = lnB + b * (CMODEL * HW);
  int s0 = (m0 & (HW - 1)) + (lane & 15);
  v8f acc0 = {}, acc1 = {}, acc2 = {}, acc3 = {};
  for (int k0 = 0; k0 < K; k0 += 32) {
    v16h bf = load_b_frag(Wc, K, col, k0, lane);
    v16h a0 = load_a_strided(A0 + s0,      A1 + s0,      k0, lane);
    v16h a1 = load_a_strided(A0 + s0 + 16, A1 + s0 + 16, k0, lane);
    v16h a2 = load_a_strided(A0 + s0 + 32, A1 + s0 + 32, k0, lane);
    v16h a3 = load_a_strided(A0 + s0 + 48, A1 + s0 + 48, k0, lane);
    acc0 = WMMA_F16(a0, bf, acc0);
    acc1 = WMMA_F16(a1, bf, acc1);
    acc2 = WMMA_F16(a2, bf, acc2);
    acc3 = WMMA_F16(a3, bf, acc3);
  }
  int roff = (lane & 16) ? 8 : 0;
  float bv = bias[col];
  float* ob = out + b * (CMODEL * HW) + col * HW;
  const float* rb = resid + b * (CMODEL * HW) + col * HW;
  int sb = m0 & (HW - 1);
#pragma unroll
  for (int r = 0; r < 8; ++r) {
    int ss = sb + roff + r;
    ob[ss]      = acc0[r] + bv + rb[ss];
    ob[ss + 16] = acc1[r] + bv + rb[ss + 16];
    ob[ss + 32] = acc2[r] + bv + rb[ss + 32];
    ob[ss + 48] = acc3[r] + bv + rb[ss + 48];
  }
}

// ---------------- layernorm over contiguous 96-wide rows (one wave32 per row) ----------------
__global__ void layernorm96(const float* __restrict__ x, float* __restrict__ o,
                            const float* __restrict__ w, const float* __restrict__ b,
                            int rows, float eps) {
  int lane = threadIdx.x & 31;
  int row = blockIdx.x * (blockDim.x >> 5) + (threadIdx.x >> 5);
  if (row >= rows) return;
  const float* xr = x + row * CMODEL;
  float v0 = xr[lane], v1 = xr[lane + 32], v2 = xr[lane + 64];
  float s = v0 + v1 + v2;
#pragma unroll
  for (int m = 16; m; m >>= 1) s += __shfl_xor(s, m, 32);
  float mu = s * (1.0f / 96.0f);
  float d0 = v0 - mu, d1 = v1 - mu, d2 = v2 - mu;
  float vs = d0 * d0 + d1 * d1 + d2 * d2;
#pragma unroll
  for (int m = 16; m; m >>= 1) vs += __shfl_xor(vs, m, 32);
  float inv = rsqrtf(vs * (1.0f / 96.0f) + eps);
  float* orow = o + row * CMODEL;
  orow[lane]      = d0 * inv * w[lane]      + b[lane];
  orow[lane + 32] = d1 * inv * w[lane + 32] + b[lane + 32];
  orow[lane + 64] = d2 * inv * w[lane + 64] + b[lane + 64];
}

// ---------------- causal depthwise conv1d + SiLU, and dt/dA precompute ----------------
__global__ void conv_dt_kernel(const float* __restrict__ proj,     // NTOK x DPROJ
                               const float* __restrict__ convw, const float* __restrict__ convb,
                               const float* __restrict__ dt_bias, const float* __restrict__ A_log,
                               float* __restrict__ convx,          // NTOK x CONVDIM (post-SiLU)
                               float* __restrict__ dt, float* __restrict__ dA) {
  int idx = blockIdx.x * blockDim.x + threadIdx.x;   // token*240 + c
  int token = idx / (CONVDIM + NHEADS);
  int c = idx - token * (CONVDIM + NHEADS);
  if (token >= NTOK) return;
  int j = token & (HW - 1);                          // position within batch sequence
  if (c < CONVDIM) {
    float acc = 0.f;
#pragma unroll
    for (int k = 0; k < DCONV; ++k) {
      int tj = j - 3 + k;
      float xv = (tj >= 0) ? proj[(token - 3 + k) * DPROJ + DINNER + c] : 0.f;
      acc += convw[c * DCONV + k] * xv;
    }
    acc += convb[c];
    convx[token * CONVDIM + c] = acc / (1.f + expf(-acc));   // SiLU
  } else {
    int h = c - CONVDIM;
    float raw = proj[token * DPROJ + DINNER + CONVDIM + h] + dt_bias[h];
    float sp = (raw > 20.f) ? raw : log1pf(expf(raw));       // softplus
    dt[token * NHEADS + h] = sp;
    dA[token * NHEADS + h] = expf(-expf(A_log[h]) * sp);
  }
}

// ---------------- chunked scan, pass 1: per-chunk end state S and decay product P ----------
__global__ void scan_chunk_state(const float* __restrict__ convx, const float* __restrict__ dt,
                                 const float* __restrict__ dA,
                                 float* __restrict__ S, float* __restrict__ P) {
  __shared__ float sB[DSTATE], sdA[NHEADS], sdt[NHEADS];
  int chunk = blockIdx.x;                // 0..127 (batch-aligned: 64 chunks per batch)
  int tid = threadIdx.x;                 // 0..191
  int h = tid / HEADDIM, p = tid - h * HEADDIM;
  int t0 = chunk * LC;
  float s[DSTATE];
#pragma unroll
  for (int n = 0; n < DSTATE; ++n) s[n] = 0.f;
  float prod = 1.f;
  for (int t = 0; t < LC; ++t) {
    int m = t0 + t;
    __syncthreads();
    if (tid < DSTATE) {
      sB[tid]  = convx[m * CONVDIM + DINNER + tid];
      sdA[tid] = dA[m * NHEADS + tid];
      sdt[tid] = dt[m * NHEADS + tid];
    }
    __syncthreads();
    float dAv = sdA[h];
    float cc  = sdt[h] * convx[m * CONVDIM + h * HEADDIM + p];
    prod *= dAv;
#pragma unroll
    for (int n = 0; n < DSTATE; ++n) s[n] = dAv * s[n] + cc * sB[n];
  }
  float* Sout = S + (chunk * NHEADS + h) * (HEADDIM * DSTATE) + p * DSTATE;
#pragma unroll
  for (int n = 0; n < DSTATE; ++n) Sout[n] = s[n];
  if (p == 0) P[chunk * NHEADS + h] = prod;
}

// ---------------- pass 2: inter-chunk prefix -> chunk-initial states ----------------
__global__ void scan_chunk_prefix(const float* __restrict__ S, const float* __restrict__ P,
                                  float* __restrict__ Hinit) {
  int bh = blockIdx.x;                   // b*NHEADS + h, 32 blocks
  int b = bh / NHEADS, h = bh - b * NHEADS;
  int tid = threadIdx.x;                 // p*16+n, 192 threads
  float hs = 0.f;
  for (int c = 0; c < NCHUNK / BATCH; ++c) {
    int chunk = b * (NCHUNK / BATCH) + c;
    int off = (chunk * NHEADS + h) * (HEADDIM * DSTATE) + tid;
    Hinit[off] = hs;
    hs = P[chunk * NHEADS + h] * hs + S[off];
  }
}

// ---------------- pass 3: intra-chunk re-scan from Hinit, emit y ----------------
__global__ void scan_output(const float* __restrict__ convx, const float* __restrict__ dt,
                            const float* __restrict__ dA, const float* __restrict__ Hinit,
                            const float* __restrict__ Dh, float* __restrict__ y) {
  __shared__ float sB[DSTATE], sC[DSTATE], sdA[NHEADS], sdt[NHEADS];
  int chunk = blockIdx.x;
  int tid = threadIdx.x;
  int h = tid / HEADDIM, p = tid - h * HEADDIM;
  int t0 = chunk * LC;
  float s[DSTATE];
  const float* hi = Hinit + (chunk * NHEADS + h) * (HEADDIM * DSTATE) + p * DSTATE;
#pragma unroll
  for (int n = 0; n < DSTATE; ++n) s[n] = hi[n];
  float dhv = Dh[h];
  for (int t = 0; t < LC; ++t) {
    int m = t0 + t;
    __syncthreads();
    if (tid < DSTATE) {
      sB[tid]  = convx[m * CONVDIM + DINNER + tid];
      sC[tid]  = convx[m * CONVDIM + DINNER + DSTATE + tid];
      sdA[tid] = dA[m * NHEADS + tid];
      sdt[tid] = dt[m * NHEADS + tid];
    }
    __syncthreads();
    float xv  = convx[m * CONVDIM + h * HEADDIM + p];
    float dAv = sdA[h];
    float cc  = sdt[h] * xv;
    float acc = dhv * xv;
#pragma unroll
    for (int n = 0; n < DSTATE; ++n) { s[n] = dAv * s[n] + cc * sB[n]; acc += s[n] * sC[n]; }
    y[m * DINNER + h * HEADDIM + p] = acc;
  }
}

// ---------------- gate with SiLU(z), RMS-norm over 192, scale by rms_w (in place) --------
__global__ void gate_rms(float* __restrict__ y, const float* __restrict__ z, int zld,
                         const float* __restrict__ rms_w) {
  int lane = threadIdx.x & 31;
  int row = blockIdx.x * (blockDim.x >> 5) + (threadIdx.x >> 5);
  if (row >= NTOK) return;
  float* yr = y + row * DINNER;
  const float* zr = z + row * zld;
  float g[6]; float ss = 0.f;
#pragma unroll
  for (int i = 0; i < 6; ++i) {
    int c = lane + i * 32;
    float zv = zr[c];
    float gv = yr[c] * (zv / (1.f + expf(-zv)));
    g[i] = gv; ss += gv * gv;
  }
#pragma unroll
  for (int m = 16; m; m >>= 1) ss += __shfl_xor(ss, m, 32);
  float inv = rsqrtf(ss * (1.0f / (float)DINNER) + 1e-5f);
#pragma unroll
  for (int i = 0; i < 6; ++i) { int c = lane + i * 32; yr[c] = g[i] * inv * rms_w[c]; }
}

// ---------------- host-side orchestration ----------------
static inline int gemm_grid64(int M, int N) {        // 64x16 supertiles, 8 waves/block
  int tiles = (M >> 6) * (N >> 4);
  return (tiles + 7) / 8;
}

extern "C" void kernel_launch(void* const* d_in, const int* in_sizes, int n_in,
                              void* d_out, int out_size, void* d_ws, size_t ws_size,
                              hipStream_t stream) {
  const float* input   = (const float*)d_in[0];
  const float* t_in    = (const float*)d_in[1];
  const float* ln1_w   = (const float*)d_in[2];
  const float* ln1_b   = (const float*)d_in[3];
  const float* Win     = (const float*)d_in[4];   // 432 x 96
  const float* convw   = (const float*)d_in[5];   // 224 x 4
  const float* convb   = (const float*)d_in[6];
  const float* dt_bias = (const float*)d_in[7];
  const float* A_log   = (const float*)d_in[8];
  const float* Dh      = (const float*)d_in[9];
  const float* rms_w   = (const float*)d_in[10];
  const float* Wout    = (const float*)d_in[11];  // 96 x 192
  const float* norm_w  = (const float*)d_in[12];
  const float* norm_b  = (const float*)d_in[13];
  const float* conv2dw = (const float*)d_in[14];  // 96 x 192
  const float* conv2db = (const float*)d_in[15];
  float* out = (float*)d_out;

  float* ws = (float*)d_ws;
  size_t o = 0;
  float* xn    = ws + o; o += (size_t)2 * NTOK * CMODEL;     // xn rows 0..8191, iln rows 8192..
  float* proj  = ws + o; o += (size_t)2 * NTOK * DPROJ;      // projX then projI
  float* convx = ws + o; o += (size_t)NTOK * CONVDIM;
  float* dt    = ws + o; o += (size_t)NTOK * NHEADS;
  float* dAb   = ws + o; o += (size_t)NTOK * NHEADS;
  float* Sbuf  = ws + o; o += (size_t)NCHUNK * NHEADS * HEADDIM * DSTATE;
  float* Pbuf  = ws + o; o += (size_t)NCHUNK * NHEADS;
  float* Hbuf  = ws + o; o += (size_t)NCHUNK * NHEADS * HEADDIM * DSTATE;
  float* ybuf  = ws + o; o += (size_t)NTOK * DINNER;
  float* z2    = ws + o; o += (size_t)NTOK * DINNER;
  float* out1  = ws + o; o += (size_t)NTOK * CMODEL;
  float* out2  = ws + o; o += (size_t)NTOK * CMODEL;
  float* lnA   = ws + o; o += (size_t)NTOK * CMODEL;
  float* lnB   = ws + o; o += (size_t)NTOK * CMODEL;
  float* iln   = xn + (size_t)NTOK * CMODEL;
  float* projI = proj + (size_t)NTOK * DPROJ;

  const int TPB = 256;
  const int lnGrid = NTOK / 8;                    // one wave per row, 8 waves/block
  const int cdGrid = (NTOK * (CONVDIM + NHEADS) + TPB - 1) / TPB;

  // Phase A: input layernorms
  layernorm96<<<lnGrid, TPB, 0, stream>>>(input, xn,  ln1_w, ln1_b, NTOK, 1e-6f);
  layernorm96<<<lnGrid, TPB, 0, stream>>>(t_in,  iln, ln1_w, ln1_b, NTOK, 1e-6f);

  // Phase B: in-projection for both streams in one WMMA GEMM (16384 x 432, K=96)
  gemm_wmma<<<gemm_grid64(2 * NTOK, DPROJ), TPB, 0, stream>>>(xn, Win, proj,
                                                              2 * NTOK, DPROJ, CMODEL);

  // Phase C: mamba #1 (u = xn path -> projX; gate z = projI[:, :192])
  conv_dt_kernel<<<cdGrid, TPB, 0, stream>>>(proj, convw, convb, dt_bias, A_log,
                                             convx, dt, dAb);
  scan_chunk_state<<<NCHUNK, NHEADS * HEADDIM, 0, stream>>>(convx, dt, dAb, Sbuf, Pbuf);
  scan_chunk_prefix<<<BATCH * NHEADS, HEADDIM * DSTATE, 0, stream>>>(Sbuf, Pbuf, Hbuf);
  scan_output<<<NCHUNK, NHEADS * HEADDIM, 0, stream>>>(convx, dt, dAb, Hbuf, Dh, ybuf);
  gate_rms<<<lnGrid, TPB, 0, stream>>>(ybuf, projI, DPROJ, rms_w);
  gemm_wmma<<<gemm_grid64(NTOK, CMODEL), TPB, 0, stream>>>(ybuf, Wout, out1,
                                                           NTOK, CMODEL, DINNER);

  // Phase D: z2 = out1 @ Win[:192].T  (8192 x 192, K=96)
  gemm_wmma<<<gemm_grid64(NTOK, DINNER), TPB, 0, stream>>>(out1, Win, z2,
                                                           NTOK, DINNER, CMODEL);

  // Phase E: mamba #2 (u = iln path -> projI; gate z = z2)
  conv_dt_kernel<<<cdGrid, TPB, 0, stream>>>(projI, convw, convb, dt_bias, A_log,
                                             convx, dt, dAb);
  scan_chunk_state<<<NCHUNK, NHEADS * HEADDIM, 0, stream>>>(convx, dt, dAb, Sbuf, Pbuf);
  scan_chunk_prefix<<<BATCH * NHEADS, HEADDIM * DSTATE, 0, stream>>>(Sbuf, Pbuf, Hbuf);
  scan_output<<<NCHUNK, NHEADS * HEADDIM, 0, stream>>>(convx, dt, dAb, Hbuf, Dh, ybuf);
  gate_rms<<<lnGrid, TPB, 0, stream>>>(ybuf, z2, DINNER, rms_w);
  gemm_wmma<<<gemm_grid64(NTOK, CMODEL), TPB, 0, stream>>>(ybuf, Wout, out2,
                                                           NTOK, CMODEL, DINNER);

  // Phase F: final layernorms + fused concat-conv1x1 + bias + residual
  layernorm96<<<lnGrid, TPB, 0, stream>>>(out1, lnA, norm_w, norm_b, NTOK, 1e-5f);
  layernorm96<<<lnGrid, TPB, 0, stream>>>(out2, lnB, norm_w, norm_b, NTOK, 1e-5f);
  gemm_final_wmma<<<gemm_grid64(NTOK, CMODEL), TPB, 0, stream>>>(lnA, lnB, conv2dw,
                                                                 conv2db, input, out);
  (void)in_sizes; (void)n_in; (void)out_size; (void)ws_size;
}
